// EnsembleMultiHeadAttention_50405736186010
// MI455X (gfx1250) — matrix operand order repaired and checked by reading, not compiled, not run on hardware
//
#include <hip/hip_runtime.h>
#include <hip/hip_bf16.h>

// ---------------------------------------------------------------------------
// EnsembleMultiHeadAttention on MI455X (gfx1250, wave32, WMMA).
// All dense contractions go through v_wmma_f32_16x16x32_bf16 (f32 accum).
// Gate MLP stays on VALU (gather-bound, co-executes with the XDL pipe).
// ---------------------------------------------------------------------------

typedef __attribute__((ext_vector_type(16))) __bf16 v16bf;
typedef __attribute__((ext_vector_type(8)))  float  v8f;
typedef __attribute__((ext_vector_type(4)))  unsigned int u32x4;

#define DEV __device__ __forceinline__

// Problem sizes (fixed by the reference)
constexpr int Bn = 4, Ln = 256, En = 2, Dn = 512, Hn = 8, DKn = 64, KGn = 16;

// Workspace layout (bytes)
constexpr size_t OFF_QBF  = 0;          // [B,L,Eq,D] bf16        2 MB
constexpr size_t OFF_KBF  = 2u  << 20;  // [B,L,Ek,D] bf16        2 MB
constexpr size_t OFF_VBF  = 4u  << 20;  // [B,L,Ek,D] bf16        2 MB
constexpr size_t OFF_WQT  = 6u  << 20;  // [Eq,512,512] bf16 (T)  1 MB
constexpr size_t OFF_WQ2T = 7u  << 20;
constexpr size_t OFF_WKT  = 8u  << 20;
constexpr size_t OFF_WK2T = 9u  << 20;
constexpr size_t OFF_WVT  = 10u << 20;
constexpr size_t OFF_WFCT = 11u << 20;
constexpr size_t OFF_QP   = 12u << 20;  // [B,L,Eq,512] bf16      2 MB
constexpr size_t OFF_QP2  = 14u << 20;
constexpr size_t OFF_KP   = 16u << 20;
constexpr size_t OFF_KP2  = 18u << 20;
constexpr size_t OFF_VPT  = 20u << 20;  // [B,Ek,H,dv,L] bf16     2 MB
constexpr size_t OFF_PRE  = 22u << 20;  // [B,L,Eq,512] bf16      2 MB
constexpr size_t OFF_FCO  = 24u << 20;  // [B,L,Eq,512] f32       4 MB
constexpr size_t WS_NEED  = 28u << 20;

DEV unsigned short f2bf(float f) {
  union { float f; unsigned u; } x; x.f = f;
  unsigned r = x.u + 0x7FFFu + ((x.u >> 16) & 1u);
  return (unsigned short)(r >> 16);
}

DEV v8f zero8() { v8f z = {0.f,0.f,0.f,0.f,0.f,0.f,0.f,0.f}; return z; }

// A operand (16-bit 16x32): per lane one contiguous 16-value run at khalf*16.
// Caller passes p = row_base + kk + khalf*16.
DEV v16bf loadA(const unsigned short* p) {
  union { v16bf v; u32x4 q[2]; } u;
  u.q[0] = *reinterpret_cast<const u32x4*>(p);
  u.q[1] = *reinterpret_cast<const u32x4*>(p + 8);
  return u.v;
}

// B operand (16-bit 32x16): per lane one contiguous 16-value run at khalf*16.
// Caller passes p = row_base + kk + khalf*16.
DEV v16bf loadB(const unsigned short* p) {
  union { v16bf v; u32x4 q[2]; } u;
  u.q[0] = *reinterpret_cast<const u32x4*>(p);
  u.q[1] = *reinterpret_cast<const u32x4*>(p + 8);
  return u.v;
}

DEV v8f wmma_bf16(v16bf a, v16bf b, v8f c) {
  return __builtin_amdgcn_wmma_f32_16x16x32_bf16(false, a, false, b,
                                                 (short)0, c, false, false);
}

// ---------------------------------------------------------------------------
// Kernel 1: fp32 -> bf16 converts + weight transposes (WT[e][n][k] = W[e][k][n])
// ---------------------------------------------------------------------------
__global__ __launch_bounds__(256) void cvt_kernel(
    const float* __restrict__ q, const float* __restrict__ k,
    const float* __restrict__ v, const float* __restrict__ Wq,
    const float* __restrict__ Wq2, const float* __restrict__ Wk,
    const float* __restrict__ Wk2, const float* __restrict__ Wv,
    const float* __restrict__ Wfc, char* __restrict__ ws) {
  const int task = blockIdx.y;
  const int idx = blockIdx.x * 256 + threadIdx.x;
  if (task < 3) {
    const float* src = (task == 0) ? q : (task == 1) ? k : v;
    unsigned short* dst = (unsigned short*)(ws + (size_t)task * (2u << 20));
    dst[idx] = f2bf(src[idx]);
  } else {
    if (idx >= En * Dn * Dn) return;  // 524288
    const float* src; size_t off;
    switch (task) {
      case 3: src = Wq;  off = OFF_WQT;  break;
      case 4: src = Wq2; off = OFF_WQ2T; break;
      case 5: src = Wk;  off = OFF_WKT;  break;
      case 6: src = Wk2; off = OFF_WK2T; break;
      case 7: src = Wv;  off = OFF_WVT;  break;
      default: src = Wfc; off = OFF_WFCT; break;
    }
    unsigned short* dst = (unsigned short*)(ws + off);
    const int e = idx >> 18, rem = idx & 262143, n = rem >> 9, kd = rem & 511;
    dst[idx] = f2bf(src[(size_t)e * 262144 + (size_t)kd * 512 + n]);
  }
}

// ---------------------------------------------------------------------------
// Kernel 2: projections qp/qp2/kp/kp2/vp. One 16x16 tile per wave, K=512.
// vp is written transposed: vpT[b][j][h][d][l].
// ---------------------------------------------------------------------------
__global__ __launch_bounds__(128) void proj_kernel(char* __restrict__ ws) {
  const int z = blockIdx.z, p = z >> 1, e = z & 1;
  const int wv = threadIdx.x >> 5, lane = threadIdx.x & 31;
  const int m = lane & 15, kh = lane >> 4;
  const int mt = blockIdx.x;                // 0..63  (M = B*L = 1024)
  const int nt = blockIdx.y * 4 + wv;       // 0..31  (N = 512)

  const size_t srcOff[5] = {OFF_QBF, OFF_QBF, OFF_KBF, OFF_KBF, OFF_VBF};
  const size_t wOff[5]   = {OFF_WQT, OFF_WQ2T, OFF_WKT, OFF_WK2T, OFF_WVT};
  const size_t dOff[5]   = {OFF_QP, OFF_QP2, OFF_KP, OFF_KP2, OFF_VPT};

  const unsigned short* A  = (const unsigned short*)(ws + srcOff[p]);
  const unsigned short* Wt = (const unsigned short*)(ws + wOff[p]) + (size_t)e * 262144;
  unsigned short* dst = (unsigned short*)(ws + dOff[p]);

  const int row = mt * 16 + m;                                  // b*L + l
  const unsigned short* arow = A + ((size_t)row * En + e) * Dn;
  const unsigned short* brow = Wt + (size_t)(nt * 16 + m) * Dn;

  v8f acc = zero8();
#pragma unroll 4
  for (int kk = 0; kk < Dn; kk += 32)
    acc = wmma_bf16(loadA(arow + kk + kh * 16), loadB(brow + kk + kh * 16), acc);

  const int col = nt * 16 + m;
#pragma unroll
  for (int r = 0; r < 8; ++r) {
    const int grow = mt * 16 + kh * 8 + r;
    const unsigned short bv = f2bf(acc[r]);
    if (p < 4) {
      dst[((size_t)grow * En + e) * Dn + col] = bv;
    } else {  // vpT[b][j][h][d][l]
      const int b = grow >> 8, l = grow & 255, hh = col >> 6, d = col & 63;
      dst[((size_t)((b * En + e) * Hn + hh) * DKn + d) * Ln + l] = bv;
    }
  }
}

// ---------------------------------------------------------------------------
// Kernel 3: fused scores + adj-mix + gate MLP + log-sigmoid + mask + joint
// softmax(512) + attn*V.  One block per (b, i, h, 32-row q tile), 8 waves.
// ---------------------------------------------------------------------------
__global__ __launch_bounds__(256) void attn_kernel(
    const unsigned short* __restrict__ qp, const unsigned short* __restrict__ qp2,
    const unsigned short* __restrict__ kp, const unsigned short* __restrict__ kp2,
    const unsigned short* __restrict__ vpT, unsigned short* __restrict__ preBuf,
    const int* __restrict__ mask, const float* __restrict__ grm,
    const float* __restrict__ adj, const float* __restrict__ gW1,
    const float* __restrict__ gb1, const float* __restrict__ gW2,
    const float* __restrict__ gb2) {
  __shared__ float sS[32][512];             // 64 KB scores
  __shared__ unsigned short sA[32][512];    // 32 KB exp(scores) bf16
  __shared__ float sRed[32][8];
  __shared__ float sMax[32];
  __shared__ float sSum[32];
  __shared__ float sW1[2][16][16];
  __shared__ float sW2[2][16];
  __shared__ float sB1[2][16];
  __shared__ float sB2[2];

  const int tid = threadIdx.x;
  const int wv = tid >> 5, lane = tid & 31;
  const int m = lane & 15, kh = lane >> 4;
  const int bz = blockIdx.z, b = bz >> 1, i = bz & 1;
  const int h = blockIdx.y;
  const int q0 = blockIdx.x * 32;

  // stage gate weights (for this (i,h), both j) in LDS
  for (int idx = tid; idx < 512; idx += 256) {
    const int j = idx >> 8, rem = idx & 255, g = rem >> 4, mm = rem & 15;
    sW1[j][g][mm] = gW1[((size_t)((i * 2 + j) * Hn + h) * 16 + g) * 16 + mm];
  }
  if (tid < 32) {
    const int j = tid >> 4, mm = tid & 15;
    sW2[j][mm] = gW2[(size_t)((i * 2 + j) * Hn + h) * 16 + mm];
    sB1[j][mm] = gb1[(size_t)((i * 2 + j) * Hn + h) * 16 + mm];
  }
  if (tid < 2) sB2[tid] = gb2[(i * 2 + tid) * Hn + h];
  __syncthreads();

  const float inv_scale = 0.125f;  // 1/sqrt(dk)

  // ---- score phase: wave w covers key blocks {w*16, w*16+128} x j in {0,1}
  for (int mt = 0; mt < 2; ++mt) {
    for (int pb = 0; pb < 2; ++pb) {
      const int l0 = wv * 16 + pb * 128;
      v8f s1a[2], s2a[2];
#pragma unroll
      for (int j = 0; j < 2; ++j) {
        v8f a1 = zero8(), a2 = zero8();
        const int qrow = q0 + mt * 16 + m;
        const unsigned short* q1r = qp  + ((size_t)(b * Ln + qrow) * En + i) * Dn + h * DKn;
        const unsigned short* q2r = qp2 + ((size_t)(b * Ln + qrow) * En + i) * Dn + h * DKn;
        const int krow = l0 + m;
        const unsigned short* k1r = kp  + ((size_t)(b * Ln + krow) * En + j) * Dn + h * DKn;
        const unsigned short* k2r = kp2 + ((size_t)(b * Ln + krow) * En + j) * Dn + h * DKn;
#pragma unroll
        for (int kk = 0; kk < DKn; kk += 32) {
          a1 = wmma_bf16(loadA(q1r + kk + kh * 16), loadB(k1r + kk + kh * 16), a1);
          a2 = wmma_bf16(loadA(q2r + kk + kh * 16), loadB(k2r + kk + kh * 16), a2);
        }
        s1a[j] = a1; s2a[j] = a2;
      }
      // epilogue: adj mix + gate MLP + log-sigmoid + mask  (t loads shared by j)
#pragma unroll
      for (int r = 0; r < 8; ++r) {
        const int qg = q0 + mt * 16 + kh * 8 + r;
        const int lg = l0 + m;
        const float at = (lg <= qg) ? adj[(size_t)(b * Ln + qg) * Ln + lg] : 0.0f;
        const int mk = mask[(size_t)(b * Ln + qg) * Ln + lg];
        float t[16];
#pragma unroll
        for (int g = 0; g < 16; ++g)
          t[g] = grm[((size_t)(b * KGn + g) * Ln + qg) * Ln + lg];
#pragma unroll
        for (int j = 0; j < 2; ++j) {
          float s = (s1a[j][r] * at + s2a[j][r] * (1.0f - at)) * inv_scale;
          float agg = sB2[j];
#pragma unroll
          for (int mm = 0; mm < 16; ++mm) {
            float hs = sB1[j][mm];
#pragma unroll
            for (int g = 0; g < 16; ++g) hs = fmaf(t[g], sW1[j][g][mm], hs);
            agg = fmaf(fmaxf(hs, 0.0f), sW2[j][mm], agg);
          }
          const float gated = (mk == 1) ? agg : 0.0f;
          s += fminf(gated, 0.0f) - log1pf(__expf(-fabsf(gated)));
          if (mk == 0) s = -1e9f;
          sS[qg - q0][j * 256 + lg] = s;
        }
      }
    }
  }
  __syncthreads();

  // ---- joint softmax over 512 columns (8 threads per row)
  {
    const int row = tid >> 3, seg = tid & 7;
    float mx = -3.0e38f;
    for (int c = seg * 64; c < seg * 64 + 64; ++c) mx = fmaxf(mx, sS[row][c]);
    sRed[row][seg] = mx;
    __syncthreads();
    if (seg == 0) {
      float vm = sRed[row][0];
      for (int s2 = 1; s2 < 8; ++s2) vm = fmaxf(vm, sRed[row][s2]);
      sMax[row] = vm;
    }
    __syncthreads();
    const float rm = sMax[row];
    float sum = 0.0f;
    for (int c = seg * 64; c < seg * 64 + 64; ++c) {
      const float ev = __expf(sS[row][c] - rm);
      sum += ev;
      sA[row][c] = f2bf(ev);  // unnormalized; 1/rowsum folded into PV epilogue
    }
    sRed[row][seg] = sum;
    __syncthreads();
    if (seg == 0) {
      float vs = 0.0f;
      for (int s2 = 0; s2 < 8; ++s2) vs += sRed[row][s2];
      sSum[row] = vs;
    }
    __syncthreads();
  }

  // ---- PV: pre[q,d] = (1/sum_q) * sum_c exp[q,c] * vpT[.,d,c]
  {
    const int mt = wv >> 2, ntd = wv & 3;
    const int d = ntd * 16 + m;
    const unsigned short* arow = &sA[mt * 16 + m][0];
    v8f acc = zero8();
#pragma unroll 4
    for (int kk = 0; kk < 512; kk += 32) {
      const int c0 = kk + kh * 16;            // contiguous 16-col run, single j
      const int j0 = c0 >> 8, key0 = c0 & 255;
      acc = wmma_bf16(loadA(arow + kk + kh * 16),
                      loadB(vpT + ((size_t)((b * En + j0) * Hn + h) * DKn + d) * Ln + key0),
                      acc);
    }
#pragma unroll
    for (int r = 0; r < 8; ++r) {
      const int qr = mt * 16 + kh * 8 + r;
      const float val = acc[r] / sSum[qr];
      const int qg = q0 + qr;
      preBuf[((size_t)(b * Ln + qg) * En + i) * Dn + h * DKn + d] = f2bf(val);
    }
  }
}

// ---------------------------------------------------------------------------
// Kernel 4: out = pre x Wfc + q (residual), f32 output
// ---------------------------------------------------------------------------
__global__ __launch_bounds__(128) void fc_kernel(
    const unsigned short* __restrict__ pre, const unsigned short* __restrict__ WfcT,
    const float* __restrict__ qin, float* __restrict__ fco) {
  const int wv = threadIdx.x >> 5, lane = threadIdx.x & 31;
  const int m = lane & 15, kh = lane >> 4;
  const int mt = blockIdx.x, nt = blockIdx.y * 4 + wv, e = blockIdx.z;
  const unsigned short* arow = pre + ((size_t)(mt * 16 + m) * En + e) * Dn;
  const unsigned short* brow = WfcT + ((size_t)e * Dn + nt * 16 + m) * Dn;
  v8f acc = zero8();
#pragma unroll 4
  for (int kk = 0; kk < Dn; kk += 32)
    acc = wmma_bf16(loadA(arow + kk + kh * 16), loadB(brow + kk + kh * 16), acc);
  const int col = nt * 16 + m;
#pragma unroll
  for (int r = 0; r < 8; ++r) {
    const int grow = mt * 16 + kh * 8 + r;
    const size_t off = ((size_t)grow * En + e) * Dn + col;
    fco[off] = acc[r] + qin[off];
  }
}

// ---------------------------------------------------------------------------
// Kernel 5: LayerNorm over D=512 per row
// ---------------------------------------------------------------------------
__global__ __launch_bounds__(256) void ln_kernel(
    const float* __restrict__ x, const float* __restrict__ g,
    const float* __restrict__ bta, float* __restrict__ out) {
  __shared__ float rs[256], rq[256];
  const int tid = threadIdx.x;
  const size_t base = (size_t)blockIdx.x * Dn;
  const float v0 = x[base + tid], v1 = x[base + tid + 256];
  rs[tid] = v0 + v1;
  rq[tid] = v0 * v0 + v1 * v1;
  __syncthreads();
  for (int st = 128; st > 0; st >>= 1) {
    if (tid < st) { rs[tid] += rs[tid + st]; rq[tid] += rq[tid + st]; }
    __syncthreads();
  }
  const float mean = rs[0] * (1.0f / Dn);
  const float var = rq[0] * (1.0f / Dn) - mean * mean;
  const float inv = rsqrtf(var + 1e-6f);
  out[base + tid] = (v0 - mean) * inv * g[tid] + bta[tid];
  out[base + tid + 256] = (v1 - mean) * inv * g[tid + 256] + bta[tid + 256];
}

// ---------------------------------------------------------------------------
extern "C" void kernel_launch(void* const* d_in, const int* in_sizes, int n_in,
                              void* d_out, int out_size, void* d_ws, size_t ws_size,
                              hipStream_t stream) {
  (void)in_sizes; (void)n_in; (void)out_size;
  if (ws_size < WS_NEED) return;

  const float* q    = (const float*)d_in[0];
  const float* k    = (const float*)d_in[1];
  const float* v    = (const float*)d_in[2];
  const int*   mask = (const int*)d_in[3];
  const float* grm  = (const float*)d_in[4];
  const float* adj  = (const float*)d_in[5];
  const float* Wq   = (const float*)d_in[6];
  const float* Wk   = (const float*)d_in[7];
  const float* Wv   = (const float*)d_in[8];
  const float* Wq2  = (const float*)d_in[9];
  const float* Wk2  = (const float*)d_in[10];
  // d_in[11] (Wv2) is computed-but-unused in the reference; skipped.
  const float* Wfc  = (const float*)d_in[12];
  const float* gW1  = (const float*)d_in[13];
  const float* gb1  = (const float*)d_in[14];
  const float* gW2  = (const float*)d_in[15];
  const float* gb2  = (const float*)d_in[16];
  const float* lng  = (const float*)d_in[17];
  const float* lnb  = (const float*)d_in[18];
  char* ws = (char*)d_ws;

  cvt_kernel<<<dim3(4096, 9), dim3(256), 0, stream>>>(q, k, v, Wq, Wq2, Wk, Wk2,
                                                      Wv, Wfc, ws);
  proj_kernel<<<dim3(64, 8, 10), dim3(128), 0, stream>>>(ws);
  attn_kernel<<<dim3(8, 8, 8), dim3(256), 0, stream>>>(
      (const unsigned short*)(ws + OFF_QP),  (const unsigned short*)(ws + OFF_QP2),
      (const unsigned short*)(ws + OFF_KP),  (const unsigned short*)(ws + OFF_KP2),
      (const unsigned short*)(ws + OFF_VPT), (unsigned short*)(ws + OFF_PRE),
      mask, grm, adj, gW1, gb1, gW2, gb2);
  fc_kernel<<<dim3(64, 8, 2), dim3(128), 0, stream>>>(
      (const unsigned short*)(ws + OFF_PRE), (const unsigned short*)(ws + OFF_WFCT),
      q, (float*)(ws + OFF_FCO));
  ln_kernel<<<dim3(2048), dim3(256), 0, stream>>>(
      (const float*)(ws + OFF_FCO), lng, lnb, (float*)d_out);
}
